// MolGNN_43078521979121
// MI455X (gfx1250) — compile-verified, matrix operand order checked by reading, not run on hardware
//
#include <hip/hip_runtime.h>

typedef unsigned int  uint32;
typedef unsigned short ushort16;
typedef __attribute__((ext_vector_type(16))) __bf16 v16bf;
typedef __attribute__((ext_vector_type(8)))  float  v8f;
typedef __attribute__((ext_vector_type(8)))  uint32 v8u;
typedef __attribute__((ext_vector_type(2)))  uint32 v2u;

#define NNODES 300000
#define NEDGES 600000
#define NGRAPH 10000
#define HID    256
#define OUTD   768
#define BN_EPS 1e-5f

__device__ __forceinline__ ushort16 f2bf(float f) {
    uint32 u = __builtin_bit_cast(uint32, f);
    u += 0x7FFFu + ((u >> 16) & 1u);           // round-to-nearest-even
    return (ushort16)(u >> 16);
}

// ---------------------------------------------------------------------------
// Pack a row-major fp32 weight [Kdim, Ndim] into bf16 WMMA B-fragment order.
// tile (kt, nt); per lane (half = lane>>4, n = nt*16 + lane&15); 8 dword
// pairs p, covering K = kmap(p,half)+{0,1}, kmap = p<4 ? half*8+2p
// : 16+half*8+2(p-4)  (mirror of the documented 16-bit A VGPR layout).
// ---------------------------------------------------------------------------
__global__ __launch_bounds__(256) void pack_b(const float* __restrict__ W,
                                              ushort16* __restrict__ P,
                                              int Kdim, int Ndim) {
    int t = blockIdx.x * 256 + threadIdx.x;
    if (t >= Kdim * Ndim) return;
    int q    = t & 1;
    int p    = (t >> 1) & 7;
    int lane = (t >> 4) & 31;
    int tile = t >> 9;
    int ntiles = Ndim >> 4;
    int nt = tile % ntiles;
    int kt = tile / ntiles;
    int half = lane >> 4;
    int n = (nt << 4) + (lane & 15);
    int k = (kt << 5) + ((p < 4) ? (half * 8 + 2 * p) : (16 + half * 8 + 2 * (p - 4))) + q;
    P[t] = f2bf(W[(size_t)k * Ndim + n]);
}

// ---------------------------------------------------------------------------
// Embedding encoders: h[n,c] = sum_f emb[f, x[n,f], c]
// ---------------------------------------------------------------------------
__global__ __launch_bounds__(256) void atom_embed(const int* __restrict__ x,
                                                  const float* __restrict__ emb,
                                                  float* __restrict__ h) {
    int n = blockIdx.x, c = threadIdx.x;
    float acc = 0.f;
#pragma unroll
    for (int f = 0; f < 9; ++f) {
        int idx = x[n * 9 + f];                          // uniform -> scalar load
        acc += emb[((size_t)(f * 119 + idx)) * HID + c];
    }
    h[(size_t)n * HID + c] = acc;
}

__global__ __launch_bounds__(256) void bond_embed(const int* __restrict__ ea,
                                                  const float* __restrict__ emb,
                                                  float* __restrict__ e) {
    int n = blockIdx.x, c = threadIdx.x;
    float acc = 0.f;
#pragma unroll
    for (int f = 0; f < 3; ++f) {
        int idx = ea[n * 3 + f];
        acc += emb[((size_t)(f * 22 + idx)) * HID + c];
    }
    e[(size_t)n * HID + c] = acc;
}

// z = (1 + eps[layer]) * h   (float4 per thread)
__global__ __launch_bounds__(256) void zinit(const float* __restrict__ h,
                                             const float* __restrict__ eps, int layer,
                                             float* __restrict__ z) {
    size_t i = ((size_t)blockIdx.x * 256 + threadIdx.x);
    float s = 1.f + eps[layer];
    float4 v = ((const float4*)h)[i];
    v.x *= s; v.y *= s; v.z *= s; v.w *= s;
    ((float4*)z)[i] = v;
}

// msg = relu(h[src] + e); scatter-add into z[dst]  (4 channels / thread)
__global__ __launch_bounds__(256) void message(const float* __restrict__ h,
                                               const float* __restrict__ e,
                                               const int* __restrict__ ei,
                                               float* __restrict__ z) {
    size_t gid = (size_t)blockIdx.x * 256 + threadIdx.x;   // NEDGES*64 threads
    int edge = (int)(gid >> 6);
    int c4   = (int)(gid & 63) << 2;
    int src = ei[edge];
    int dst = ei[NEDGES + edge];
    float4 hv = *(const float4*)(h + (size_t)src * HID + c4);
    float4 ev = *(const float4*)(e + (size_t)edge * HID + c4);
    float* zp = z + (size_t)dst * HID + c4;
    __hip_atomic_fetch_add(zp + 0, fmaxf(hv.x + ev.x, 0.f), __ATOMIC_RELAXED, __HIP_MEMORY_SCOPE_AGENT);
    __hip_atomic_fetch_add(zp + 1, fmaxf(hv.y + ev.y, 0.f), __ATOMIC_RELAXED, __HIP_MEMORY_SCOPE_AGENT);
    __hip_atomic_fetch_add(zp + 2, fmaxf(hv.z + ev.z, 0.f), __ATOMIC_RELAXED, __HIP_MEMORY_SCOPE_AGENT);
    __hip_atomic_fetch_add(zp + 3, fmaxf(hv.w + ev.w, 0.f), __ATOMIC_RELAXED, __HIP_MEMORY_SCOPE_AGENT);
}

// global_add_pool: g[batch[n]] += h[n]
__global__ __launch_bounds__(256) void pool(const float* __restrict__ h,
                                            const int* __restrict__ bids,
                                            float* __restrict__ g) {
    size_t gid = (size_t)blockIdx.x * 256 + threadIdx.x;   // NNODES*64 threads
    int n  = (int)(gid >> 6);
    int c4 = (int)(gid & 63) << 2;
    int b = bids[n];
    float4 hv = *(const float4*)(h + (size_t)n * HID + c4);
    float* gp = g + (size_t)b * HID + c4;
    __hip_atomic_fetch_add(gp + 0, hv.x, __ATOMIC_RELAXED, __HIP_MEMORY_SCOPE_AGENT);
    __hip_atomic_fetch_add(gp + 1, hv.y, __ATOMIC_RELAXED, __HIP_MEMORY_SCOPE_AGENT);
    __hip_atomic_fetch_add(gp + 2, hv.z, __ATOMIC_RELAXED, __HIP_MEMORY_SCOPE_AGENT);
    __hip_atomic_fetch_add(gp + 3, hv.w, __ATOMIC_RELAXED, __HIP_MEMORY_SCOPE_AGENT);
}

__global__ __launch_bounds__(256) void zerof(float* __restrict__ p, int n4) {
    int i = blockIdx.x * 256 + threadIdx.x;
    if (i < n4) ((float4*)p)[i] = make_float4(0.f, 0.f, 0.f, 0.f);
}

// ---------------------------------------------------------------------------
// WMMA GEMM: C[M,Ndim] = A[M,KD] @ Bpacked + fused epilogue.
// Block = 256 threads = 8 waves; block computes a 16 x 256 output slab
// (each wave owns two 16x16 N-tiles). The FULL 16xKD A slab is staged into
// LDS once (bf16, padded row stride KD/2+2 dwords -> aligned stores,
// near-conflict-free fragment reads), so the fully unrolled K-loop is a pure
// stream of B-loads + ds gathers + v_wmma with a single barrier pair.
// bf16 A sources are staged with GLOBAL_LOAD_ASYNC_TO_LDS_B128 (ASYNCcnt);
// fp32 A sources are converted to bf16 in VALU on the way in.
// EP: 1 = BN+ReLU -> bf16 | 2 = BN+ReLU+residual -> f32
//     3 = bias+ReLU -> bf16 | 4 = bias -> f32
// ---------------------------------------------------------------------------
template <int EP, bool ABF16, int KD>
__global__ __launch_bounds__(256) void gemm_wmma(
    const void* __restrict__ Asrc, const ushort16* __restrict__ Bp,
    const int Ndim,
    const float* __restrict__ e0, const float* __restrict__ e1,
    const float* __restrict__ e2, const float* __restrict__ e3,
    const float* __restrict__ e4,
    const float* __restrict__ resid,
    float* __restrict__ out32, ushort16* __restrict__ out16) {
    constexpr int Kd2  = KD / 2;          // dwords (bf16 pairs) per A row
    constexpr int LSTR = Kd2 + 2;         // padded LDS row stride in dwords
    constexpr int CH   = Kd2 / 16;        // dwords staged per thread
    __shared__ uint32 aLds[16 * LSTR];

    const int tid  = threadIdx.x;
    const int wave = tid >> 5;
    const int lane = tid & 31;
    const int m0   = blockIdx.x << 4;
    const int n0w  = (blockIdx.y << 8) + (wave << 5);
    const int ntiles = Ndim >> 4;
    const int nt0  = n0w >> 4;

    // ---- stage the whole 16 x KD A slab into LDS as bf16
    {
        const int srow = tid >> 4;                 // 0..15
        const int scol = (tid & 15) * CH;          // dword column
        uint32* ldsRow = aLds + srow * LSTR + scol;
        if (ABF16) {
            const char* gsrc = (const char*)Asrc +
                (((size_t)(m0 + srow) * KD + (size_t)scol * 2) * 2);
#pragma unroll
            for (int j = 0; j < CH / 4; ++j) {     // 16B per async op
                uint32 loff = (uint32)(size_t)(void*)(ldsRow + j * 4);
                const void* gp = gsrc + j * 16;
                asm volatile("global_load_async_to_lds_b128 %0, %1, off"
                             :: "v"(loff), "v"(gp) : "memory");
            }
        } else {
            const float* A32 = (const float*)Asrc +
                (size_t)(m0 + srow) * KD + scol * 2;
#pragma unroll
            for (int j = 0; j < CH / 2; ++j) {
                const float4 fv = *(const float4*)(A32 + j * 4);
                v2u pr = { (uint32)f2bf(fv.x) | ((uint32)f2bf(fv.y) << 16),
                           (uint32)f2bf(fv.z) | ((uint32)f2bf(fv.w) << 16) };
                *(v2u*)(ldsRow + j * 2) = pr;
            }
        }
    }
    if (ABF16) asm volatile("s_wait_asynccnt 0" ::: "memory");
    __syncthreads();

    // ---- fully unrolled K-loop: 2 WMMAs per k-step
    v8f c0 = {0.f, 0.f, 0.f, 0.f, 0.f, 0.f, 0.f, 0.f};
    v8f c1 = {0.f, 0.f, 0.f, 0.f, 0.f, 0.f, 0.f, 0.f};
    const int mrow  = lane & 15;
    const int h4    = (lane >> 4) << 2;
    const int abase = mrow * LSTR;
#pragma unroll
    for (int kt = 0; kt < KD / 32; ++kt) {
        const v8u bu0 = *(const v8u*)(Bp + ((((size_t)kt * ntiles + nt0    ) * 32 + lane) << 4));
        const v8u bu1 = *(const v8u*)(Bp + ((((size_t)kt * ntiles + nt0 + 1) * 32 + lane) << 4));
        v8u au;
#pragma unroll
        for (int p = 0; p < 8; ++p) {
            const int kd = (p < 4) ? (h4 + p) : (8 + h4 + (p - 4));
            au[p] = aLds[abase + kt * 16 + kd];
        }
        const v16bf a = __builtin_bit_cast(v16bf, au);
        c0 = __builtin_amdgcn_wmma_f32_16x16x32_bf16(false, a, false,
                 __builtin_bit_cast(v16bf, bu0), (short)0, c0, false, false);
        c1 = __builtin_amdgcn_wmma_f32_16x16x32_bf16(false, a, false,
                 __builtin_bit_cast(v16bf, bu1), (short)0, c1, false, false);
    }

    // ---- fused epilogue; C/D layout: VGPR r -> M = r + 8*(lane>>4), N = lane&15
    const int half8 = (lane >> 4) << 3;
#pragma unroll
    for (int j = 0; j < 2; ++j) {
        const v8f cc = j ? c1 : c0;
        const int n = n0w + (j << 4) + mrow;
        float scale = 1.f, bias;
        if (EP == 1 || EP == 2) {
            scale = e0[n] * rsqrtf(e1[n] + BN_EPS);
            bias  = (e2[n] - e3[n]) * scale + e4[n];
        } else {
            bias = e0[n];
        }
#pragma unroll
        for (int r = 0; r < 8; ++r) {
            const size_t idx = (size_t)(m0 + half8 + r) * Ndim + n;
            float y = cc[r];
            if (EP == 1)      { y = fmaxf(y * scale + bias, 0.f); out16[idx] = f2bf(y); }
            else if (EP == 2) { y = fmaxf(y * scale + bias, 0.f); out32[idx] = resid[idx] + y; }
            else if (EP == 3) { y = fmaxf(y + bias, 0.f);         out16[idx] = f2bf(y); }
            else              {                                    out32[idx] = y + bias; }
        }
    }
}

// Row-wise L2 normalize of d_out [NGRAPH, 768] in place.
__global__ __launch_bounds__(256) void l2norm(float* __restrict__ out) {
    __shared__ float red[256];
    const int gph = blockIdx.x, t = threadIdx.x;
    float v[3], s = 0.f;
#pragma unroll
    for (int j = 0; j < 3; ++j) {
        v[j] = out[(size_t)gph * OUTD + t + j * 256];
        s += v[j] * v[j];
    }
    red[t] = s;
    __syncthreads();
    for (int o = 128; o > 0; o >>= 1) {
        if (t < o) red[t] += red[t + o];
        __syncthreads();
    }
    const float inv = 1.f / fmaxf(sqrtf(red[0]), 1e-12f);
#pragma unroll
    for (int j = 0; j < 3; ++j)
        out[(size_t)gph * OUTD + t + j * 256] = v[j] * inv;
}

extern "C" void kernel_launch(void* const* d_in, const int* in_sizes, int n_in,
                              void* d_out, int out_size, void* d_ws, size_t ws_size,
                              hipStream_t stream) {
    const int*   x     = (const int*)  d_in[0];
    const int*   eattr = (const int*)  d_in[1];
    const int*   eidx  = (const int*)  d_in[2];
    const int*   bids  = (const int*)  d_in[3];
    const float* aemb  = (const float*)d_in[4];
    const float* bemb  = (const float*)d_in[5];
    const float* W1    = (const float*)d_in[6];
    const float* b1    = (const float*)d_in[7];
    const float* g1    = (const float*)d_in[8];
    const float* be1   = (const float*)d_in[9];
    const float* m1    = (const float*)d_in[10];
    const float* v1    = (const float*)d_in[11];
    const float* W2    = (const float*)d_in[12];
    const float* b2    = (const float*)d_in[13];
    const float* eps   = (const float*)d_in[14];
    const float* gbn   = (const float*)d_in[15];
    const float* bbn   = (const float*)d_in[16];
    const float* mbn   = (const float*)d_in[17];
    const float* vbn   = (const float*)d_in[18];
    const float* P1    = (const float*)d_in[19];
    const float* pb1   = (const float*)d_in[20];
    const float* P2    = (const float*)d_in[21];
    const float* pb2   = (const float*)d_in[22];

    // ---- workspace carve-up (all offsets 256B-aligned)
    char* w = (char*)d_ws;
    float*    h   = (float*)w;     w += (size_t)NNODES * HID * 4;       // 307.2 MB
    float*    e   = (float*)w;     w += (size_t)NEDGES * HID * 4;       // 614.4 MB
    float*    z   = (float*)w;     w += (size_t)NNODES * HID * 4;       // 307.2 MB
    ushort16* t   = (ushort16*)w;  w += (size_t)NNODES * 2 * HID * 2;   // 307.2 MB
    ushort16* q   = (ushort16*)w;  w += (size_t)NGRAPH * HID * 2;       // 5.12 MB
    float*    g   = (float*)w;     w += (size_t)NGRAPH * HID * 4;       // 10.24 MB
    ushort16* W1p = (ushort16*)w;  w += (size_t)4 * HID * 2 * HID * 2;  // 1 MB
    ushort16* W2p = (ushort16*)w;  w += (size_t)4 * 2 * HID * HID * 2;  // 1 MB
    ushort16* P1p = (ushort16*)w;  w += (size_t)HID * HID * 2;
    ushort16* P2p = (ushort16*)w;  w += (size_t)HID * OUTD * 2;

    // ---- one-time weight packing into WMMA B-fragment bf16 layout
    for (int i = 0; i < 4; ++i) {
        pack_b<<<(HID * 2 * HID + 255) / 256, 256, 0, stream>>>(
            W1 + (size_t)i * HID * 2 * HID, W1p + (size_t)i * HID * 2 * HID, HID, 2 * HID);
        pack_b<<<(2 * HID * HID + 255) / 256, 256, 0, stream>>>(
            W2 + (size_t)i * 2 * HID * HID, W2p + (size_t)i * 2 * HID * HID, 2 * HID, HID);
    }
    pack_b<<<(HID * HID + 255) / 256, 256, 0, stream>>>(P1, P1p, HID, HID);
    pack_b<<<(HID * OUTD + 255) / 256, 256, 0, stream>>>(P2, P2p, HID, OUTD);

    // ---- encoders
    atom_embed<<<NNODES, 256, 0, stream>>>(x, aemb, h);
    bond_embed<<<NEDGES, 256, 0, stream>>>(eattr, bemb, e);

    // ---- GINE layers
    for (int i = 0; i < 4; ++i) {
        zinit<<<NNODES * HID / 4 / 256, 256, 0, stream>>>(h, eps, i, z);
        message<<<NEDGES * 64 / 256, 256, 0, stream>>>(h, e, eidx, z);

        // z[N,256] @ W1 -> BN -> ReLU -> t bf16 [N,512]
        gemm_wmma<1, false, HID><<<dim3(NNODES / 16, (2 * HID) / 256), 256, 0, stream>>>(
            z, W1p + (size_t)i * HID * 2 * HID, 2 * HID,
            g1 + i * 2 * HID, v1 + i * 2 * HID, b1 + i * 2 * HID,
            m1 + i * 2 * HID, be1 + i * 2 * HID, nullptr, nullptr, t);

        // t[N,512] @ W2 -> BN -> ReLU -> h += (residual, in place)
        gemm_wmma<2, true, 2 * HID><<<dim3(NNODES / 16, HID / 256), 256, 0, stream>>>(
            t, W2p + (size_t)i * 2 * HID * HID, HID,
            gbn + i * HID, vbn + i * HID, b2 + i * HID,
            mbn + i * HID, bbn + i * HID, h, h, nullptr);
    }

    // ---- readout
    zerof<<<(NGRAPH * HID / 4 + 255) / 256, 256, 0, stream>>>(g, NGRAPH * HID / 4);
    pool<<<NNODES * 64 / 256, 256, 0, stream>>>(h, bids, g);

    // g[G,256] @ P1 -> +bias -> ReLU -> q bf16
    gemm_wmma<3, false, HID><<<dim3(NGRAPH / 16, HID / 256), 256, 0, stream>>>(
        g, P1p, HID, pb1, nullptr, nullptr, nullptr, nullptr,
        nullptr, nullptr, q);

    // q[G,256] @ P2 -> +bias -> d_out f32
    gemm_wmma<4, true, HID><<<dim3(NGRAPH / 16, OUTD / 256), 256, 0, stream>>>(
        q, P2p, OUTD, pb2, nullptr, nullptr, nullptr, nullptr,
        nullptr, (float*)d_out, nullptr);

    l2norm<<<NGRAPH, 256, 0, stream>>>((float*)d_out);
}
